// GroupedQueryAttention_80736795230469
// MI455X (gfx1250) — matrix operand order repaired and checked by reading, not compiled, not run on hardware
//
#include <hip/hip_runtime.h>
#include <hip/hip_bf16.h>
#include <math.h>

// ---------------------------------------------------------------------------
// GQA attention layer for MI455X (gfx1250, wave32, WMMA).
// All GEMMs use v_wmma_f32_16x16x32_bf16 (bf16 in, f32 accumulate).
// GEMM waves compute 32(M) x 64(N) tiles: every B-fragment feeds 2 WMMAs.
// ---------------------------------------------------------------------------

typedef __attribute__((ext_vector_type(16))) __bf16 v16bf;
typedef __attribute__((ext_vector_type(8)))  __bf16 v8bf;
typedef __attribute__((ext_vector_type(8)))  float  v8f;

union V16 { v16bf v; v8bf h[2]; };

#define T_SEQ  2048
#define DMODEL 2048
#define NH     32
#define NKV    8
#define HD     64

// ---------------------------------------------------------------------------
// f32 -> bf16 conversion, 8 elements per lane (b128-granular traffic)
// ---------------------------------------------------------------------------
__global__ void cvt_f32_to_bf16(const float* __restrict__ in,
                                __bf16* __restrict__ out, int n8) {
  int i = blockIdx.x * blockDim.x + threadIdx.x;
  if (i < n8) {
    v8f v = *(const v8f*)(in + (size_t)i * 8);
    v8bf o;
#pragma unroll
    for (int e = 0; e < 8; ++e) o[e] = (__bf16)v[e];
    *(v8bf*)(out + (size_t)i * 8) = o;
  }
}

// ---------------------------------------------------------------------------
// WMMA GEMM: out[M,N] = A[M,K] @ W[N,K]^T, K = DMODEL = 2048, M = 4096.
// Block = 128 threads = 4 waves; each wave computes a 32(M) x 64(N) tile.
// Grid = (N/64, M/128).
// mode 0: store f32 to outF                          (output projection)
// mode 1: RMSNorm + RoPE, store bf16 (B,NH ,T,64)    (Q)
// mode 2: RMSNorm + RoPE, store bf16 (B,NKV,T,64)    (K)
// mode 3: store bf16 transposed (B,NKV,64,T)         (V)
// ---------------------------------------------------------------------------
__global__ __launch_bounds__(128)
void wmma_gemm(const __bf16* __restrict__ A, const __bf16* __restrict__ W,
               int N, int mode, float* __restrict__ outF,
               __bf16* __restrict__ outB, const float* __restrict__ normw) {
  const int lane = threadIdx.x & 31;
  const int wave = threadIdx.x >> 5;
  const int n16  = lane & 15;
  const int hl   = lane >> 4;            // lane-half (0/1)
  const int m0   = (blockIdx.y * 4 + wave) * 32;
  const int n0   = blockIdx.x * 64;
  const int K    = DMODEL;
  const int ka   = hl * 16;

  v8f acc[2][4] = {};

  const __bf16* arow0 = A + (size_t)(m0 + n16) * K;
  const __bf16* arow1 = A + (size_t)(m0 + 16 + n16) * K;
  for (int k0 = 0; k0 < K; k0 += 32) {
    V16 af0, af1;
    af0.h[0] = *(const v8bf*)(arow0 + k0 + ka);
    af0.h[1] = *(const v8bf*)(arow0 + k0 + ka + 8);
    af1.h[0] = *(const v8bf*)(arow1 + k0 + ka);
    af1.h[1] = *(const v8bf*)(arow1 + k0 + ka + 8);
#pragma unroll
    for (int a = 0; a < 4; ++a) {
      const __bf16* brow = W + (size_t)(n0 + a * 16 + n16) * K + k0 + hl * 16;
      v16bf bf = *(const v16bf*)brow;
      acc[0][a] = __builtin_amdgcn_wmma_f32_16x16x32_bf16(
          false, af0.v, false, bf, (short)0, acc[0][a], false, false);
      acc[1][a] = __builtin_amdgcn_wmma_f32_16x16x32_bf16(
          false, af1.v, false, bf, (short)0, acc[1][a], false, false);
    }
  }

  if (mode == 0) {
#pragma unroll
    for (int mi = 0; mi < 2; ++mi)
#pragma unroll
      for (int j = 0; j < 8; ++j) {
        int r = m0 + mi * 16 + j + hl * 8;
#pragma unroll
        for (int a = 0; a < 4; ++a)
          outF[(size_t)r * N + n0 + a * 16 + n16] = acc[mi][a][j];
      }
    return;
  }

  if (mode == 1 || mode == 2) {
    // --- fused per-head RMSNorm (row = 16 lanes of one half) + RoPE -------
    float wgt[4];
#pragma unroll
    for (int a = 0; a < 4; ++a) wgt[a] = normw[a * 16 + n16];
    float invf[2];
#pragma unroll
    for (int a01 = 0; a01 < 2; ++a01)
      invf[a01] =
          __expf((float)(a01 * 16 + n16) * -0.28782313662425572f); // ln(1e4)/32
#pragma unroll
    for (int mi = 0; mi < 2; ++mi) {
#pragma unroll
      for (int j = 0; j < 8; ++j) {
        float ss = 0.f;
#pragma unroll
        for (int a = 0; a < 4; ++a) ss += acc[mi][a][j] * acc[mi][a][j];
        ss += __shfl_xor(ss, 1, 32);
        ss += __shfl_xor(ss, 2, 32);
        ss += __shfl_xor(ss, 4, 32);
        ss += __shfl_xor(ss, 8, 32);
        float rstd = rsqrtf(ss * (1.0f / 64.0f) + 1e-6f);
#pragma unroll
        for (int a = 0; a < 4; ++a) acc[mi][a][j] *= rstd * wgt[a];

        // RoPE: pairs (d, d+32) are (acc[a], acc[a+2]) at same lane/element
        int r = m0 + mi * 16 + j + hl * 8;
        int t = r & (T_SEQ - 1);
#pragma unroll
        for (int a01 = 0; a01 < 2; ++a01) {
          float ang = (float)t * invf[a01];
          float s, c;
          __sincosf(ang, &s, &c);
          float x1 = acc[mi][a01][j], x2 = acc[mi][a01 + 2][j];
          acc[mi][a01][j]     = x1 * c - x2 * s;
          acc[mi][a01 + 2][j] = x2 * c + x1 * s;
        }
      }
    }
    const int h      = blockIdx.x;               // N-tile == one head
    const int nheads = (mode == 1) ? NH : NKV;
#pragma unroll
    for (int mi = 0; mi < 2; ++mi)
#pragma unroll
      for (int j = 0; j < 8; ++j) {
        int r = m0 + mi * 16 + j + hl * 8;
        int b = r >> 11, t = r & (T_SEQ - 1);
        size_t base = (((size_t)b * nheads + h) * T_SEQ + t) * HD;
#pragma unroll
        for (int a = 0; a < 4; ++a)
          outB[base + a * 16 + n16] = (__bf16)acc[mi][a][j];
      }
    return;
  }

  // mode 3: V, stored transposed (B, NKV, 64, T)
  {
    const int h = blockIdx.x;
#pragma unroll
    for (int mi = 0; mi < 2; ++mi)
#pragma unroll
      for (int j = 0; j < 8; ++j) {
        int r = m0 + mi * 16 + j + hl * 8;
        int b = r >> 11, t = r & (T_SEQ - 1);
#pragma unroll
        for (int a = 0; a < 4; ++a)
          outB[(((size_t)b * NKV + h) * HD + a * 16 + n16) * T_SEQ + t] =
              (__bf16)acc[mi][a][j];
      }
  }
}

// ---------------------------------------------------------------------------
// Flash attention. Block = 4 waves; each wave owns 16 queries of one (b,h).
// Key tiles of 64 (4 WMMA N-subtiles). Online softmax, O accumulated in f32.
// Q: (B,NH,T,64) bf16   K: (B,NKV,T,64) bf16   Vt: (B,NKV,64,T) bf16
// Out: (B,T,NH*64) bf16 (feeds output projection GEMM)
// ---------------------------------------------------------------------------
__global__ __launch_bounds__(128)
void flash_attn(const __bf16* __restrict__ Q, const __bf16* __restrict__ Kc,
                const __bf16* __restrict__ Vt, __bf16* __restrict__ Oo) {
  __shared__ __align__(16) __bf16 lds_p[4][16 * 64];

  const int lane = threadIdx.x & 31;
  const int wave = threadIdx.x >> 5;
  const int n16  = lane & 15;
  const int hl   = lane >> 4;
  const int ka   = hl * 16;
  const int b    = blockIdx.z;
  const int h    = blockIdx.y;
  const int hk   = h >> 2;                       // 32 Q heads -> 8 KV heads
  const int q0   = blockIdx.x * 64 + wave * 16;

  const __bf16* qb = Q  + (((size_t)b * NH  + h ) * T_SEQ + q0) * HD;
  const __bf16* kb = Kc + (((size_t)b * NKV + hk) * T_SEQ) * HD;
  const __bf16* vb = Vt + (((size_t)b * NKV + hk) * HD) * T_SEQ;

  // Q A-fragments for the two 32-wide K-dim steps (d = 0..31, 32..63)
  V16 qf[2];
#pragma unroll
  for (int d0 = 0; d0 < 2; ++d0) {
    const __bf16* p = qb + (size_t)n16 * HD + d0 * 32;
    qf[d0].h[0] = *(const v8bf*)(p + ka);
    qf[d0].h[1] = *(const v8bf*)(p + ka + 8);
  }

  v8f O[4] = {};
  float mrow[8], lrow[8];
#pragma unroll
  for (int j = 0; j < 8; ++j) { mrow[j] = -3.0e38f; lrow[j] = 0.f; }

  const int ntiles = blockIdx.x + 1;             // causal: keys <= q0+15
  for (int kt = 0; kt < ntiles; ++kt) {
    const int kbase = kt * 64;

    // ---- S = Q @ K^T (16 x 64) -----------------------------------------
    v8f S[4];
#pragma unroll
    for (int a = 0; a < 4; ++a) {
      const __bf16* kr = kb + (size_t)(kbase + a * 16 + n16) * HD;
      v16bf bk0 = *(const v16bf*)(kr + hl * 16);
      v16bf bk1 = *(const v16bf*)(kr + 32 + hl * 16);
      v8f z = {};
      z = __builtin_amdgcn_wmma_f32_16x16x32_bf16(
          false, qf[0].v, false, bk0, (short)0, z, false, false);
      z = __builtin_amdgcn_wmma_f32_16x16x32_bf16(
          false, qf[1].v, false, bk1, (short)0, z, false, false);
      S[a] = z;
    }

    // ---- scale, causal mask, online softmax ----------------------------
#pragma unroll
    for (int j = 0; j < 8; ++j) {
      const int tq = q0 + j + hl * 8;
      float mx = -3.0e38f;
#pragma unroll
      for (int a = 0; a < 4; ++a) {
        int kc = kbase + a * 16 + n16;
        float s = S[a][j] * 0.125f;                // 1/sqrt(64)
        if (kc > tq) s = -3.0e38f;
        S[a][j] = s;
        mx = fmaxf(mx, s);
      }
      mx = fmaxf(mx, __shfl_xor(mx, 1, 32));
      mx = fmaxf(mx, __shfl_xor(mx, 2, 32));
      mx = fmaxf(mx, __shfl_xor(mx, 4, 32));
      mx = fmaxf(mx, __shfl_xor(mx, 8, 32));
      float mnew = fmaxf(mrow[j], mx);
      float rs = 0.f;
#pragma unroll
      for (int a = 0; a < 4; ++a) {
        float p = __expf(S[a][j] - mnew);
        S[a][j] = p;
        rs += p;
      }
      rs += __shfl_xor(rs, 1, 32);
      rs += __shfl_xor(rs, 2, 32);
      rs += __shfl_xor(rs, 4, 32);
      rs += __shfl_xor(rs, 8, 32);
      float alpha = __expf(mrow[j] - mnew);
      lrow[j] = lrow[j] * alpha + rs;
      mrow[j] = mnew;
#pragma unroll
      for (int a = 0; a < 4; ++a) O[a][j] *= alpha;
      // stash P row (C-layout -> LDS row-major 16x64 bf16)
#pragma unroll
      for (int a = 0; a < 4; ++a)
        lds_p[wave][(j + hl * 8) * 64 + a * 16 + n16] = (__bf16)S[a][j];
    }
    asm volatile("s_wait_dscnt 0" ::: "memory");

    // ---- reload P as A-fragments, O += P @ V ---------------------------
    V16 pf[2];
#pragma unroll
    for (int k0 = 0; k0 < 2; ++k0) {
      const __bf16* pr = &lds_p[wave][n16 * 64 + k0 * 32];
      pf[k0].h[0] = *(const v8bf*)(pr + ka);
      pf[k0].h[1] = *(const v8bf*)(pr + ka + 8);
    }
#pragma unroll
    for (int a = 0; a < 4; ++a) {
      const __bf16* vr =
          vb + (size_t)(a * 16 + n16) * T_SEQ + kbase + hl * 16;
      v16bf v0 = *(const v16bf*)vr;
      v16bf v1 = *(const v16bf*)(vr + 32);
      O[a] = __builtin_amdgcn_wmma_f32_16x16x32_bf16(
          false, pf[0].v, false, v0, (short)0, O[a], false, false);
      O[a] = __builtin_amdgcn_wmma_f32_16x16x32_bf16(
          false, pf[1].v, false, v1, (short)0, O[a], false, false);
    }
  }

  // ---- normalize and store as (B, T, NH*64) bf16 ------------------------
#pragma unroll
  for (int j = 0; j < 8; ++j) {
    float inv = 1.0f / lrow[j];
    int tq = q0 + j + hl * 8;
    size_t base = (((size_t)b * T_SEQ + tq) * DMODEL) + (size_t)h * HD;
#pragma unroll
    for (int a = 0; a < 4; ++a)
      Oo[base + a * 16 + n16] = (__bf16)(O[a][j] * inv);
  }
}

// ---------------------------------------------------------------------------
// Workspace layout (bytes)
// ---------------------------------------------------------------------------
#define OFF_X  ((size_t)0)          // x  bf16: 4096x2048      = 16 MB
#define OFF_WQ ((size_t)16777216)   // wq bf16: 2048x2048      =  8 MB
#define OFF_WK ((size_t)25165824)   // wk bf16: 512x2048       =  2 MB
#define OFF_WV ((size_t)27262976)   // wv bf16: 512x2048       =  2 MB
#define OFF_WO ((size_t)29360128)   // wo bf16: 2048x2048      =  8 MB
#define OFF_Q  ((size_t)37748736)   // q  bf16 (B,32,T,64)     = 16 MB
#define OFF_K  ((size_t)54525952)   // k  bf16 (B,8,T,64)      =  4 MB
#define OFF_V  ((size_t)58720256)   // vT bf16 (B,8,64,T)      =  4 MB
#define OFF_A  ((size_t)62914560)   // attn bf16 (B*T, 2048)   = 16 MB

extern "C" void kernel_launch(void* const* d_in, const int* in_sizes, int n_in,
                              void* d_out, int out_size, void* d_ws,
                              size_t ws_size, hipStream_t stream) {
  const float* x   = (const float*)d_in[0];
  const float* wq  = (const float*)d_in[1];
  const float* wk  = (const float*)d_in[2];
  const float* wv  = (const float*)d_in[3];
  const float* wo  = (const float*)d_in[4];
  const float* qnw = (const float*)d_in[5];
  const float* knw = (const float*)d_in[6];

  char* ws = (char*)d_ws;
  __bf16* xb  = (__bf16*)(ws + OFF_X);
  __bf16* wqb = (__bf16*)(ws + OFF_WQ);
  __bf16* wkb = (__bf16*)(ws + OFF_WK);
  __bf16* wvb = (__bf16*)(ws + OFF_WV);
  __bf16* wob = (__bf16*)(ws + OFF_WO);
  __bf16* qws = (__bf16*)(ws + OFF_Q);
  __bf16* kws = (__bf16*)(ws + OFF_K);
  __bf16* vws = (__bf16*)(ws + OFF_V);
  __bf16* aws = (__bf16*)(ws + OFF_A);

  // 1) bf16 conversions (8 elements / lane)
  cvt_f32_to_bf16<<<(1048576 + 255) / 256, 256, 0, stream>>>(x,  xb,  1048576);
  cvt_f32_to_bf16<<<( 524288 + 255) / 256, 256, 0, stream>>>(wq, wqb,  524288);
  cvt_f32_to_bf16<<<( 131072 + 255) / 256, 256, 0, stream>>>(wk, wkb,  131072);
  cvt_f32_to_bf16<<<( 131072 + 255) / 256, 256, 0, stream>>>(wv, wvb,  131072);
  cvt_f32_to_bf16<<<( 524288 + 255) / 256, 256, 0, stream>>>(wo, wob,  524288);

  // 2) QKV projections with fused RMSNorm/RoPE (Q,K) and transpose (V)
  wmma_gemm<<<dim3(32, 32), 128, 0, stream>>>(xb, wqb, 2048, 1, nullptr, qws, qnw);
  wmma_gemm<<<dim3(8,  32), 128, 0, stream>>>(xb, wkb,  512, 2, nullptr, kws, knw);
  wmma_gemm<<<dim3(8,  32), 128, 0, stream>>>(xb, wvb,  512, 3, nullptr, vws, nullptr);

  // 3) causal flash attention, GQA 4:1
  flash_attn<<<dim3(T_SEQ / 64, NH, 2), 128, 0, stream>>>(qws, kws, vws, aws);

  // 4) output projection -> f32
  wmma_gemm<<<dim3(32, 32), 128, 0, stream>>>(aws, wob, 2048, 0, (float*)d_out,
                                              nullptr, nullptr);
}
